// ParallelMultiHeadAttention_13280038879618
// MI455X (gfx1250) — matrix operand order repaired and checked by reading, not compiled, not run on hardware
//
#include <hip/hip_runtime.h>

// ---------------------------------------------------------------------------
// Types for CDNA5 WMMA (gfx1250, wave32): 16x16x32 bf16 -> f32
// ---------------------------------------------------------------------------
typedef __attribute__((ext_vector_type(16))) __bf16 v16bf;
typedef __attribute__((ext_vector_type(8)))  float  v8f;

struct __align__(16) U4 { unsigned int x, y, z, w; };

union FragU {
    v16bf f;
    U4 q[2];
    unsigned short us[16];
};

// gfx1250 async global->LDS copy path (ASYNCcnt-tracked), with sync fallback.
#if defined(__has_builtin)
#if __has_builtin(__builtin_amdgcn_global_load_async_to_lds_b128) && \
    __has_builtin(__builtin_amdgcn_s_wait_asynccnt)
#define USE_ASYNC_LDS 1
#endif
#endif

#if defined(USE_ASYNC_LDS)
typedef int vi4 __attribute__((vector_size(16)));
typedef __attribute__((address_space(1))) vi4* gvi4p;   // global-AS int4*
typedef __attribute__((address_space(3))) vi4* lvi4p;   // LDS-AS int4*

__device__ __forceinline__ void async_copy16(const unsigned short* g,
                                             unsigned short* l) {
    __builtin_amdgcn_global_load_async_to_lds_b128(
        (gvi4p)(vi4*)(void*)g, (lvi4p)(vi4*)(void*)l, 0, 0);
}
#endif

__device__ __forceinline__ unsigned short f2bf(float x) {
    unsigned int u = __float_as_uint(x);
    u += 0x7fffu + ((u >> 16) & 1u);          // round-to-nearest-even
    return (unsigned short)(u >> 16);
}

// ---------------------------------------------------------------------------
// f32 -> bf16 conversion kernel
// ---------------------------------------------------------------------------
__global__ void cvt_bf16_k(const float* __restrict__ in,
                           unsigned short* __restrict__ out, int n) {
    int i = blockIdx.x * blockDim.x + threadIdx.x;
    if (i < n) out[i] = f2bf(in[i]);
}

// ---------------------------------------------------------------------------
// GEMM: C[M=8192, N=1024] = A[M,K=1024](bf16) * B[N,K](bf16)^T + bias
// Block tile 128x128, 8 waves (4 M-groups x 2 N-groups), wave tile 32x64
// (2x4 WMMA). Double-buffered LDS; async global->LDS prefetch of tile i+1
// overlapped with WMMA on tile i.
// mode 0: f32 row-major out                       (final projection, +b_o)
// mode 1: bf16 out at [b,h,l,d], f = d*H + h      (Q, K split)
// mode 2: bf16 out at [b,h,d,l]                   (transposed V)
// ---------------------------------------------------------------------------
__global__ __launch_bounds__(256) void gemm_bf16_k(
    const unsigned short* __restrict__ A,
    const unsigned short* __restrict__ Bw,
    const float* __restrict__ bias,
    void* __restrict__ Cout, int K, int mode)
{
    __shared__ unsigned short As[2][128 * 32];
    __shared__ unsigned short Bs[2][128 * 32];

    const int tid  = threadIdx.x;
    const int w    = tid >> 5, lane = tid & 31;
    const int hl   = lane >> 4, ln = lane & 15;
    const int wm   = (w & 3) * 32, wn = (w >> 2) * 64;
    const int rowBase = blockIdx.y * 128, colBase = blockIdx.x * 128;

    // Per-thread staging slots: 2 A chunks + 2 B chunks of 16B per K-step.
    const int srow = tid >> 2, scc = (tid & 3) * 8;

    auto stage = [&](int kk, int buf) {
#pragma unroll
        for (int t = 0; t < 2; ++t) {
            int row = srow + t * 64;
            const unsigned short* ga = &A[(size_t)(rowBase + row) * K + kk + scc];
            const unsigned short* gb = &Bw[(size_t)(colBase + row) * K + kk + scc];
            unsigned short* la = &As[buf][row * 32 + scc];
            unsigned short* lb = &Bs[buf][row * 32 + scc];
#if defined(USE_ASYNC_LDS)
            async_copy16(ga, la);
            async_copy16(gb, lb);
#else
            *(U4*)la = *(const U4*)ga;
            *(U4*)lb = *(const U4*)gb;
#endif
        }
    };

    v8f acc[2][4];
#pragma unroll
    for (int j = 0; j < 4; ++j) {
        float bv = bias[colBase + wn + j * 16 + ln];
#pragma unroll
        for (int i = 0; i < 2; ++i)
#pragma unroll
            for (int r = 0; r < 8; ++r) acc[i][j][r] = bv;
    }

    stage(0, 0);                               // prologue prefetch

    for (int kk = 0; kk < K; kk += 32) {
        const int buf = (kk >> 5) & 1;
#if defined(USE_ASYNC_LDS)
        __builtin_amdgcn_s_wait_asynccnt(0);   // my loads into buf complete
#endif
        __syncthreads();                       // everyone's loads into buf done;
                                               // all reads of buf^1 retired
        if (kk + 32 < K) stage(kk + 32, buf ^ 1);

        FragU a[2], b[4];
#pragma unroll
        for (int i = 0; i < 2; ++i) {          // A frag: K = hl*16..+15
            int row = wm + i * 16 + ln;
            a[i].q[0] = *(const U4*)&As[buf][row * 32 + hl * 16];
            a[i].q[1] = *(const U4*)&As[buf][row * 32 + hl * 16 + 8];
        }
#pragma unroll
        for (int j = 0; j < 4; ++j) {          // B frag: K = hl*16..+15
            int col = wn + j * 16 + ln;
            b[j].q[0] = *(const U4*)&Bs[buf][col * 32 + hl * 16];
            b[j].q[1] = *(const U4*)&Bs[buf][col * 32 + hl * 16 + 8];
        }
#pragma unroll
        for (int i = 0; i < 2; ++i)
#pragma unroll
            for (int j = 0; j < 4; ++j)
                acc[i][j] = __builtin_amdgcn_wmma_f32_16x16x32_bf16(
                    false, a[i].f, false, b[j].f, (short)0, acc[i][j], false, false);
    }

    // Epilogue: C layout => row = r + 8*hl, col = ln
#pragma unroll
    for (int i = 0; i < 2; ++i)
#pragma unroll
        for (int j = 0; j < 4; ++j)
#pragma unroll
            for (int r = 0; r < 8; ++r) {
                int m = rowBase + wm + i * 16 + r + 8 * hl;
                int fc = colBase + wn + j * 16 + ln;
                float v = acc[i][j][r];
                if (mode == 0) {
                    ((float*)Cout)[(size_t)m * 1024 + fc] = v;
                } else {
                    int bb = m >> 11, l = m & 2047;
                    int d = fc >> 4, h = fc & 15;        // f = d*H + h (h fastest)
                    size_t idx = (mode == 1)
                        ? ((size_t)((bb * 16 + h) * 2048 + l)) * 64 + d
                        : ((size_t)((bb * 16 + h) * 64 + d)) * 2048 + l;
                    ((unsigned short*)Cout)[idx] = f2bf(v);
                }
            }
}

// ---------------------------------------------------------------------------
// Flash attention: grid (L/64, B*H), 4 waves/block, each wave = 16 Q rows.
// Q,K: [B,H,L,64] bf16;  V: [B,H,64,L] bf16 (transposed);  AO: [B,L,1024] bf16
// ---------------------------------------------------------------------------
__global__ __launch_bounds__(128) void attn_k(
    const unsigned short* __restrict__ Q,
    const unsigned short* __restrict__ Kt,
    const unsigned short* __restrict__ Vt,
    unsigned short* __restrict__ AO)
{
    __shared__ unsigned short Pt[4 * 16 * 64];     // per-wave P tiles (bf16)

    const int tid = threadIdx.x;
    const int w   = tid >> 5, lane = tid & 31;
    const int hl  = lane >> 4, ln = lane & 15;
    const int bh  = blockIdx.y;
    const int q0  = blockIdx.x * 64 + w * 16;
    const size_t qkb = (size_t)bh * 2048 * 64;
    unsigned short* Pw = &Pt[w * 16 * 64];

    // Q fragments (A-layout), K-dim = 64 in two 32-chunks
    FragU qf[2];
    {
        const unsigned short* qp = Q + qkb + (size_t)(q0 + ln) * 64;
#pragma unroll
        for (int c = 0; c < 2; ++c) {
            qf[c].q[0] = *(const U4*)(qp + c * 32 + hl * 16);
            qf[c].q[1] = *(const U4*)(qp + c * 32 + hl * 16 + 8);
        }
    }

    v8f O[4];
    float mi[8], li[8];
#pragma unroll
    for (int t = 0; t < 4; ++t)
#pragma unroll
        for (int r = 0; r < 8; ++r) O[t][r] = 0.f;
#pragma unroll
    for (int r = 0; r < 8; ++r) { mi[r] = -3.0e38f; li[r] = 0.f; }

    for (int j = 0; j < 32; ++j) {
        const int key0 = j * 64;

        // S = Q * K^T for 4 column tiles of 16 keys
        v8f S[4];
#pragma unroll
        for (int t = 0; t < 4; ++t) {
            const unsigned short* kp = Kt + qkb + (size_t)(key0 + t * 16 + ln) * 64;
            FragU k0, k1;
            k0.q[0] = *(const U4*)(kp + hl * 16);
            k0.q[1] = *(const U4*)(kp + hl * 16 + 8);
            k1.q[0] = *(const U4*)(kp + 32 + hl * 16);
            k1.q[1] = *(const U4*)(kp + 32 + hl * 16 + 8);
            v8f z = {0.f, 0.f, 0.f, 0.f, 0.f, 0.f, 0.f, 0.f};
            v8f s = __builtin_amdgcn_wmma_f32_16x16x32_bf16(
                false, qf[0].f, false, k0.f, (short)0, z, false, false);
            s = __builtin_amdgcn_wmma_f32_16x16x32_bf16(
                false, qf[1].f, false, k1.f, (short)0, s, false, false);
            S[t] = s;
        }

        // Online softmax (scale = 1/sqrt(64) = 0.125); rows live in 16-lane groups
#pragma unroll
        for (int r = 0; r < 8; ++r) {
            float s0 = S[0][r] * 0.125f, s1 = S[1][r] * 0.125f;
            float s2 = S[2][r] * 0.125f, s3 = S[3][r] * 0.125f;
            float vmax = fmaxf(fmaxf(s0, s1), fmaxf(s2, s3));
#pragma unroll
            for (int off = 8; off >= 1; off >>= 1)
                vmax = fmaxf(vmax, __shfl_xor(vmax, off, 32));
            float nm = fmaxf(mi[r], vmax);
            float corr = __expf(mi[r] - nm);
            mi[r] = nm;
            float p0 = __expf(s0 - nm), p1 = __expf(s1 - nm);
            float p2 = __expf(s2 - nm), p3 = __expf(s3 - nm);
            int prow = (r + 8 * hl) * 64 + ln;
            Pw[prow]      = f2bf(p0);
            Pw[prow + 16] = f2bf(p1);
            Pw[prow + 32] = f2bf(p2);
            Pw[prow + 48] = f2bf(p3);
            float rs = p0 + p1 + p2 + p3;
#pragma unroll
            for (int off = 8; off >= 1; off >>= 1)
                rs += __shfl_xor(rs, off, 32);
            li[r] = li[r] * corr + rs;
#pragma unroll
            for (int t = 0; t < 4; ++t) O[t][r] *= corr;
        }
        __syncthreads();

        // Reload P in A-fragment layout from LDS
        FragU pf[2];
#pragma unroll
        for (int c = 0; c < 2; ++c) {
            pf[c].q[0] = *(const U4*)&Pw[ln * 64 + c * 32 + hl * 16];
            pf[c].q[1] = *(const U4*)&Pw[ln * 64 + c * 32 + hl * 16 + 8];
        }

        // O += P * V : V transposed => B-frag loads are contiguous in L
#pragma unroll
        for (int t = 0; t < 4; ++t) {
            const unsigned short* vp =
                Vt + (size_t)bh * 64 * 2048 + (size_t)(t * 16 + ln) * 2048 + key0;
            FragU v0, v1;
            v0.q[0] = *(const U4*)(vp + hl * 16);
            v0.q[1] = *(const U4*)(vp + hl * 16 + 8);
            v1.q[0] = *(const U4*)(vp + 32 + hl * 16);
            v1.q[1] = *(const U4*)(vp + 32 + hl * 16 + 8);
            O[t] = __builtin_amdgcn_wmma_f32_16x16x32_bf16(
                false, pf[0].f, false, v0.f, (short)0, O[t], false, false);
            O[t] = __builtin_amdgcn_wmma_f32_16x16x32_bf16(
                false, pf[1].f, false, v1.f, (short)0, O[t], false, false);
        }
    }

    // Normalize and store AO[b, l, h*64 + dv]  (merge order f = h*DV + d)
    const int b = bh >> 4, h = bh & 15;
#pragma unroll
    for (int r = 0; r < 8; ++r) {
        float inv = 1.0f / li[r];
        int l = q0 + r + 8 * hl;
        size_t base = ((size_t)(b * 2048 + l)) * 1024 + h * 64 + ln;
#pragma unroll
        for (int t = 0; t < 4; ++t)
            AO[base + t * 16] = f2bf(O[t][r] * inv);
    }
}

// ---------------------------------------------------------------------------
// Host launch
// ---------------------------------------------------------------------------
extern "C" void kernel_launch(void* const* d_in, const int* in_sizes, int n_in,
                              void* d_out, int out_size, void* d_ws, size_t ws_size,
                              hipStream_t stream) {
    (void)in_sizes; (void)n_in; (void)out_size; (void)ws_size;

    const float* x  = (const float*)d_in[0];
    const float* Wq = (const float*)d_in[1];
    const float* bq = (const float*)d_in[2];
    const float* Wk = (const float*)d_in[3];
    const float* bk = (const float*)d_in[4];
    const float* Wv = (const float*)d_in[5];
    const float* bv = (const float*)d_in[6];
    const float* Wo = (const float*)d_in[7];
    const float* bo = (const float*)d_in[8];

    const size_t NX = 8388608;   // B*L*E = 4*2048*1024
    const size_t NW = 1048576;   // 1024*1024

    char* ws = (char*)d_ws;
    size_t off = 0;
    auto take = [&](size_t bytes) {
        size_t o = off; off = (off + bytes + 255) & ~(size_t)255; return o;
    };
    unsigned short* xbf = (unsigned short*)(ws + take(NX * 2));
    unsigned short* wqb = (unsigned short*)(ws + take(NW * 2));
    unsigned short* wkb = (unsigned short*)(ws + take(NW * 2));
    unsigned short* wvb = (unsigned short*)(ws + take(NW * 2));
    unsigned short* wob = (unsigned short*)(ws + take(NW * 2));
    unsigned short* qb  = (unsigned short*)(ws + take(NX * 2));
    unsigned short* kb  = (unsigned short*)(ws + take(NX * 2));
    unsigned short* vtb = (unsigned short*)(ws + take(NX * 2));
    unsigned short* aob = (unsigned short*)(ws + take(NX * 2));

    cvt_bf16_k<<<(int)((NX + 255) / 256), 256, 0, stream>>>(x,  xbf, (int)NX);
    cvt_bf16_k<<<(int)((NW + 255) / 256), 256, 0, stream>>>(Wq, wqb, (int)NW);
    cvt_bf16_k<<<(int)((NW + 255) / 256), 256, 0, stream>>>(Wk, wkb, (int)NW);
    cvt_bf16_k<<<(int)((NW + 255) / 256), 256, 0, stream>>>(Wv, wvb, (int)NW);
    cvt_bf16_k<<<(int)((NW + 255) / 256), 256, 0, stream>>>(Wo, wob, (int)NW);

    dim3 gg(8, 64);    // N/128 x M/128
    gemm_bf16_k<<<gg, 256, 0, stream>>>(xbf, wqb, bq, qb,  1024, 1);
    gemm_bf16_k<<<gg, 256, 0, stream>>>(xbf, wkb, bk, kb,  1024, 1);
    gemm_bf16_k<<<gg, 256, 0, stream>>>(xbf, wvb, bv, vtb, 1024, 2);

    attn_k<<<dim3(32, 64), 128, 0, stream>>>(qb, kb, vtb, aob);

    gemm_bf16_k<<<gg, 256, 0, stream>>>(aob, wob, bo, d_out, 1024, 0);
}
